// ConvCombAttention_36859409334956
// MI455X (gfx1250) — compile-verified, matrix operand order checked by reading, not run on hardware
//
#include <hip/hip_runtime.h>
#include <hip/hip_bf16.h>

typedef _Float16 v8h  __attribute__((ext_vector_type(8)));
typedef _Float16 v16h __attribute__((ext_vector_type(16)));
typedef float    v8f  __attribute__((ext_vector_type(8)));
typedef __attribute__((address_space(3))) _Float16 lds_f16;

#define NS 16        // batch
#define CH 384       // channels
#define LSP 256      // 16*16 spatial tokens after downsample
#define NHEADS 12
#define HD 32
#define QKVC 1152    // 3*CH
#define BW 8         // waves per block (256 threads)
#define LDSPAD 8     // +8 halves per row -> conflict-free ds_load_b128
#define SBMAX (16*(CH+LDSPAD))

// ---------------------------------------------------------------------------
// WMMA fragment layout per CDNA5 ISA 7.12.2 (16-bit A 16x32):
// lane m = lane&15 ; half j of v16h holds K = (j&7) + 16*(j>>3) + 8*(lane>>4)
// B stored N-major x K (transposed) so its loads are identical to A's.
// ---------------------------------------------------------------------------
__device__ __forceinline__ v16h load_frag16(const _Float16* __restrict__ p) {
    v8h lo = *(const v8h*)p;         // K = base .. base+7
    v8h hi = *(const v8h*)(p + 16);  // K = base+16 .. base+23
    return __builtin_shufflevector(lo, hi, 0,1,2,3,4,5,6,7,8,9,10,11,12,13,14,15);
}

// cooperative async stage of a 16-row x K-half B panel into padded LDS
__device__ __forceinline__ void stage_B(const _Float16* __restrict__ Bt, int row0,
                                        int ldb, int K, _Float16* sB) {
    const int cpr   = K >> 3;        // 16-byte chunks per row
    const int total = 16 * cpr;
    for (int ch = threadIdx.x; ch < total; ch += blockDim.x) {
        int row = ch / cpr, cc = ch - row * cpr;
        const _Float16* g = Bt + (size_t)(row0 + row) * ldb + (cc << 3);
        unsigned loff = (unsigned)(unsigned long long)(lds_f16*)&sB[row * (K + LDSPAD) + (cc << 3)];
        asm volatile("global_load_async_to_lds_b128 %0, %1, off"
                     :: "v"(loff), "v"(g) : "memory");
    }
    asm volatile("s_wait_asynccnt 0x0" ::: "memory");
    __syncthreads();
}

// one wave computes one 16x16 tile; A from global, B panel from LDS
__device__ __forceinline__ v8f wmma_tile_ldsB(const _Float16* __restrict__ A,
                                              const _Float16* __restrict__ sB,
                                              int tm, int K, int lda) {
    const int lane = threadIdx.x & 31;
    v8f acc = {0.f,0.f,0.f,0.f,0.f,0.f,0.f,0.f};
    const _Float16* ap = A + (size_t)(tm*16 + (lane & 15)) * lda + ((lane >> 4) << 3);
    const _Float16* bp = sB + (lane & 15) * (K + LDSPAD) + ((lane >> 4) << 3);
    for (int k0 = 0; k0 < K; k0 += 32) {
        v16h a = load_frag16(ap + k0);
        v16h b = load_frag16(bp + k0);   // ds_load_b128 x2
        acc = __builtin_amdgcn_wmma_f32_16x16x32_f16(false, a, false, b,
                                                     (short)0, acc, false, false);
    }
    return acc;
}
// D element r lives at (M = tm*16 + r + 8*(lane>>4), N = tn*16 + (lane&15))

// ---------------------------------------------------------------------------
// Elementwise / reduction kernels
// ---------------------------------------------------------------------------
__global__ void k_f32_to_f16(const float* __restrict__ src, _Float16* __restrict__ dst, int n) {
    int i = blockIdx.x * blockDim.x + threadIdx.x;
    if (i < n) dst[i] = (_Float16)src[i];
}

// depthwise 7x7 stride-4 conv + folded eval-mode BN -> xd [NS][CH][256]
__global__ void k_dwconv_bn(const float* __restrict__ x, const float* __restrict__ w,
                            const float* __restrict__ g, const float* __restrict__ b,
                            const float* __restrict__ mean, const float* __restrict__ var,
                            float* __restrict__ xd) {
    int idx = blockIdx.x * blockDim.x + threadIdx.x;
    if (idx >= NS*CH*LSP) return;
    int l  = idx & 255;
    int ox = l & 15, oy = l >> 4;
    int c  = (idx >> 8) % CH;
    int n  = idx / (CH * LSP);
    const float* xp = x + ((size_t)(n*CH + c) << 12);  // 64*64
    const float* wp = w + c*49;
    float acc = 0.f;
    int iy0 = oy*4 - 3, ix0 = ox*4 - 3;
    #pragma unroll
    for (int ky = 0; ky < 7; ++ky) {
        int iy = iy0 + ky; if ((unsigned)iy >= 64u) continue;
        #pragma unroll
        for (int kx = 0; kx < 7; ++kx) {
            int ix = ix0 + kx; if ((unsigned)ix >= 64u) continue;
            acc += xp[iy*64 + ix] * wp[ky*7 + kx];
        }
    }
    float s = g[c] * rsqrtf(var[c] + 1e-5f);
    xd[idx] = acc * s + (b[c] - mean[c] * s);
}

// GroupNorm(1 group) stats per sample: stats[2n]=mu, stats[2n+1]=rsqrt(var+eps)
__global__ void k_gn_stats(const float* __restrict__ xd, float* __restrict__ stats) {
    __shared__ float ssum[256], ssq[256];
    int n = blockIdx.x, t = threadIdx.x;
    const float* p = xd + (size_t)n * (CH*LSP);
    float s = 0.f, q = 0.f;
    for (int i = t; i < CH*LSP; i += 256) { float v = p[i]; s += v; q += v*v; }
    ssum[t] = s; ssq[t] = q; __syncthreads();
    for (int o = 128; o > 0; o >>= 1) {
        if (t < o) { ssum[t] += ssum[t+o]; ssq[t] += ssq[t+o]; }
        __syncthreads();
    }
    if (t == 0) {
        float mu  = ssum[0] / (float)(CH*LSP);
        float var = ssq[0]  / (float)(CH*LSP) - mu*mu;
        stats[2*n]   = mu;
        stats[2*n+1] = rsqrtf(var + 1e-5f);
    }
}

// normalize + per-channel affine, store transposed f16: xnT[n][l][c]
__global__ void k_gn_apply(const float* __restrict__ xd, const float* __restrict__ stats,
                           const float* __restrict__ g, const float* __restrict__ b,
                           _Float16* __restrict__ xnT) {
    int idx = blockIdx.x * blockDim.x + threadIdx.x;
    if (idx >= NS*CH*LSP) return;
    int l = idx & 255;
    int c = (idx >> 8) % CH;
    int n = idx / (CH * LSP);
    float mu = stats[2*n], rs = stats[2*n+1];
    float v = (xd[idx] - mu) * rs * g[c] + b[c];
    xnT[((size_t)n*LSP + l)*CH + c] = (_Float16)v;
}

// row softmax: f32 in, f16 out
__global__ void k_softmax(const float* __restrict__ S, _Float16* __restrict__ A,
                          int R, float scale) {
    __shared__ float red[128];
    int row = blockIdx.x, t = threadIdx.x;
    const float* p = S + (size_t)row * R;
    _Float16* o = A + (size_t)row * R;
    float m = -3.4e38f;
    for (int i = t; i < R; i += 128) m = fmaxf(m, p[i] * scale);
    red[t] = m; __syncthreads();
    for (int s = 64; s > 0; s >>= 1) { if (t < s) red[t] = fmaxf(red[t], red[t+s]); __syncthreads(); }
    m = red[0]; __syncthreads();
    float sum = 0.f;
    for (int i = t; i < R; i += 128) sum += __expf(p[i] * scale - m);
    red[t] = sum; __syncthreads();
    for (int s = 64; s > 0; s >>= 1) { if (t < s) red[t] += red[t+s]; __syncthreads(); }
    float inv = 1.f / red[0];
    for (int i = t; i < R; i += 128) o[i] = (_Float16)(__expf(p[i] * scale - m) * inv);
}

// ---------------------------------------------------------------------------
// GEMM kernels: 8 waves/block, shared async-staged B panel (128M x 16N per block)
// ---------------------------------------------------------------------------
// qkv channel: W[1152][384] x xnT[n]:[256][384]^T ; route rows: q,k row-major, v transposed
__global__ void k_gemm_qkv_ch(const _Float16* __restrict__ W, const _Float16* __restrict__ xnT,
                              _Float16* __restrict__ q, _Float16* __restrict__ k,
                              _Float16* __restrict__ vT) {
    __shared__ __align__(16) _Float16 sB[SBMAX];
    int lane = threadIdx.x & 31;
    int tn = blockIdx.x & 15, tg = blockIdx.x >> 4;
    int n = blockIdx.y;
    stage_B(xnT + (size_t)n*LSP*CH, tn*16, CH, CH, sB);
    int tm = tg*BW + (threadIdx.x >> 5);
    v8f acc = wmma_tile_ldsB(W, sB, tm, CH, CH);
    int col = tn*16 + (lane & 15);
    int mbase = tm*16 + ((lane >> 4) << 3);
    #pragma unroll
    for (int r = 0; r < 8; ++r) {
        int o = mbase + r;
        _Float16 h = (_Float16)acc[r];
        if (o < CH)          q[((size_t)n*CH + o)*LSP + col] = h;
        else if (o < 2*CH)   k[((size_t)n*CH + (o - CH))*LSP + col] = h;
        else                 vT[((size_t)n*LSP + col)*CH + (o - 2*CH)] = h;
    }
}

// qkv spatial: route into per-head q[l][d], k[l][d], v[d][l]
__global__ void k_gemm_qkv_sp(const _Float16* __restrict__ W, const _Float16* __restrict__ xnT,
                              _Float16* __restrict__ q, _Float16* __restrict__ k,
                              _Float16* __restrict__ v) {
    __shared__ __align__(16) _Float16 sB[SBMAX];
    int lane = threadIdx.x & 31;
    int tn = blockIdx.x & 15, tg = blockIdx.x >> 4;
    int n = blockIdx.y;
    stage_B(xnT + (size_t)n*LSP*CH, tn*16, CH, CH, sB);
    int tm = tg*BW + (threadIdx.x >> 5);
    v8f acc = wmma_tile_ldsB(W, sB, tm, CH, CH);
    int l = tn*16 + (lane & 15);
    int mbase = tm*16 + ((lane >> 4) << 3);
    #pragma unroll
    for (int r = 0; r < 8; ++r) {
        int o = mbase + r;             // 0..1151 = h*96 + rr
        int h = o / 96, rr = o % 96;
        _Float16 hv = (_Float16)acc[r];
        size_t hb = (size_t)(n*NHEADS + h);
        if (rr < HD)          q[(hb*LSP + l)*HD + rr] = hv;
        else if (rr < 2*HD)   k[(hb*LSP + l)*HD + (rr - HD)] = hv;
        else                  v[(hb*HD + (rr - 2*HD))*LSP + l] = hv;
    }
}

// generic score GEMM: S = A (M x K) * B^T (N x K), f32 out; lda=ldb=K
__global__ void k_gemm_score(const _Float16* __restrict__ A, const _Float16* __restrict__ B,
                             float* __restrict__ S, int tilesN, int K, int Ncols,
                             int strideA, int strideB, int strideS) {
    __shared__ __align__(16) _Float16 sB[SBMAX];
    int lane = threadIdx.x & 31;
    int tn = blockIdx.x % tilesN, tg = blockIdx.x / tilesN;
    int bz = blockIdx.y;
    stage_B(B + (size_t)bz*strideB, tn*16, K, K, sB);
    int tm = tg*BW + (threadIdx.x >> 5);
    v8f acc = wmma_tile_ldsB(A + (size_t)bz*strideA, sB, tm, K, K);
    int col = tn*16 + (lane & 15);
    int mbase = tm*16 + ((lane >> 4) << 3);
    float* Sp = S + (size_t)bz*strideS;
    #pragma unroll
    for (int r = 0; r < 8; ++r)
        Sp[(size_t)(mbase + r)*Ncols + col] = acc[r];
}

// channel att*v: att[n]:[384][384] x vT[n]:[256][384]^T -> store transposed valsT[n][l][c]
__global__ void k_gemm_av_ch(const _Float16* __restrict__ att, const _Float16* __restrict__ vT,
                             _Float16* __restrict__ valsT) {
    __shared__ __align__(16) _Float16 sB[SBMAX];
    int lane = threadIdx.x & 31;
    int tn = blockIdx.x & 15, tg = blockIdx.x >> 4;
    int n = blockIdx.y;
    stage_B(vT + (size_t)n*LSP*CH, tn*16, CH, CH, sB);
    int tm = tg*BW + (threadIdx.x >> 5);
    v8f acc = wmma_tile_ldsB(att + (size_t)n*CH*CH, sB, tm, CH, CH);
    int col = tn*16 + (lane & 15);
    int mbase = tm*16 + ((lane >> 4) << 3);
    #pragma unroll
    for (int r = 0; r < 8; ++r)
        valsT[((size_t)n*LSP + col)*CH + (mbase + r)] = (_Float16)acc[r];
}

// spatial att*v per sample: att[h]:[256][256] x v[h]:[32][256]^T -> spvT[l][h*32+d]
__global__ void k_gemm_av_sp(const _Float16* __restrict__ att, const _Float16* __restrict__ v,
                             _Float16* __restrict__ outT) {
    __shared__ __align__(16) _Float16 sB[SBMAX];
    int lane = threadIdx.x & 31;
    int tn = blockIdx.x & 1, tg = blockIdx.x >> 1;
    int h = blockIdx.y;
    stage_B(v + (size_t)h*HD*LSP, tn*16, LSP, LSP, sB);
    int tm = tg*BW + (threadIdx.x >> 5);
    v8f acc = wmma_tile_ldsB(att + (size_t)h*LSP*LSP, sB, tm, LSP, LSP);
    int d = tn*16 + (lane & 15);
    int mbase = tm*16 + ((lane >> 4) << 3);
    #pragma unroll
    for (int r = 0; r < 8; ++r)
        outT[(size_t)(mbase + r)*CH + h*HD + d] = (_Float16)acc[r];
}

// out projection: W[384][384] x valsT[n]:[256][384]^T ; epilogue xd += D + bias
__global__ void k_gemm_out(const _Float16* __restrict__ W, const _Float16* __restrict__ valsT,
                           const float* __restrict__ bias, float* __restrict__ xd) {
    __shared__ __align__(16) _Float16 sB[SBMAX];
    int lane = threadIdx.x & 31;
    int tn = blockIdx.x & 15, tg = blockIdx.x >> 4;
    int n = blockIdx.y;
    stage_B(valsT + (size_t)n*LSP*CH, tn*16, CH, CH, sB);
    int tm = tg*BW + (threadIdx.x >> 5);
    v8f acc = wmma_tile_ldsB(W, sB, tm, CH, CH);
    int col = tn*16 + (lane & 15);
    int mbase = tm*16 + ((lane >> 4) << 3);
    #pragma unroll
    for (int r = 0; r < 8; ++r) {
        int m = mbase + r;
        size_t i = ((size_t)n*CH + m)*LSP + col;
        xd[i] += acc[r] + bias[m];
    }
}

// depthwise ConvTranspose2d k=8 s=4 p=2 + residual + bias
__global__ void k_upsample(const float* __restrict__ x, const float* __restrict__ xd,
                           const float* __restrict__ w, const float* __restrict__ bias,
                           float* __restrict__ out) {
    int idx = blockIdx.x * blockDim.x + threadIdx.x;
    if (idx >= NS*CH*64*64) return;
    int xx = idx & 63;
    int y  = (idx >> 6) & 63;
    int c  = (idx >> 12) % CH;
    int n  = idx / (CH * 4096);
    const float* wp = w + c*64;
    const float* xp = xd + (size_t)(n*CH + c)*LSP;
    float acc = 0.f;
    int imin = (y >= 5) ? ((y - 2) >> 2) : 0;
    int imax = (y + 2) >> 2; if (imax > 15) imax = 15;
    int jmin = (xx >= 5) ? ((xx - 2) >> 2) : 0;
    int jmax = (xx + 2) >> 2; if (jmax > 15) jmax = 15;
    for (int i = imin; i <= imax; ++i) {
        int ky = y + 2 - 4*i;          // in [0,7]
        for (int j = jmin; j <= jmax; ++j) {
            int kx = xx + 2 - 4*j;     // in [0,7]
            acc += xp[i*16 + j] * wp[ky*8 + kx];
        }
    }
    out[idx] = x[idx] + acc + bias[c];
}

// ---------------------------------------------------------------------------
extern "C" void kernel_launch(void* const* d_in, const int* in_sizes, int n_in,
                              void* d_out, int out_size, void* d_ws, size_t ws_size,
                              hipStream_t stream) {
    (void)in_sizes; (void)n_in; (void)out_size; (void)ws_size;
    const float* x        = (const float*)d_in[0];
    const float* dw_w     = (const float*)d_in[1];
    const float* bn_g     = (const float*)d_in[2];
    const float* bn_b     = (const float*)d_in[3];
    const float* bn_m     = (const float*)d_in[4];
    const float* bn_v     = (const float*)d_in[5];
    const float* lnch_g   = (const float*)d_in[6];
    const float* lnch_b   = (const float*)d_in[7];
    const float* w_qkv_ch = (const float*)d_in[8];
    const float* w_out_ch = (const float*)d_in[9];
    const float* b_out_ch = (const float*)d_in[10];
    const float* lnsp_g   = (const float*)d_in[11];
    const float* lnsp_b   = (const float*)d_in[12];
    const float* w_qkv_sp = (const float*)d_in[13];
    const float* w_out_sp = (const float*)d_in[14];
    const float* b_out_sp = (const float*)d_in[15];
    const float* up_w     = (const float*)d_in[16];
    const float* up_b     = (const float*)d_in[17];
    float* out = (float*)d_out;

    char* ws = (char*)d_ws;
    size_t off = 0;
    auto take = [&](size_t bytes) -> char* {
        char* p = ws + off;
        off = (off + bytes + 255) & ~(size_t)255;
        return p;
    };
    float*    xd    = (float*)   take((size_t)NS*CH*LSP*4);
    float*    stats = (float*)   take((size_t)NS*2*4);
    _Float16* xnT   = (_Float16*)take((size_t)NS*LSP*CH*2);
    _Float16* wq16c = (_Float16*)take((size_t)QKVC*CH*2);
    _Float16* wo16c = (_Float16*)take((size_t)CH*CH*2);
    _Float16* wq16s = (_Float16*)take((size_t)QKVC*CH*2);
    _Float16* wo16s = (_Float16*)take((size_t)CH*CH*2);
    _Float16* qch   = (_Float16*)take((size_t)NS*CH*LSP*2);
    _Float16* kch   = (_Float16*)take((size_t)NS*CH*LSP*2);
    _Float16* vTch  = (_Float16*)take((size_t)NS*LSP*CH*2);
    float*    Sch   = (float*)   take((size_t)NS*CH*CH*4);
    _Float16* attch = (_Float16*)take((size_t)NS*CH*CH*2);
    _Float16* valsT = (_Float16*)take((size_t)NS*LSP*CH*2);
    _Float16* qsp   = (_Float16*)take((size_t)NS*NHEADS*LSP*HD*2);
    _Float16* ksp   = (_Float16*)take((size_t)NS*NHEADS*LSP*HD*2);
    _Float16* vsp   = (_Float16*)take((size_t)NS*NHEADS*HD*LSP*2);
    float*    Ssp   = (float*)   take((size_t)NHEADS*LSP*LSP*4);   // per-sample
    _Float16* attsp = (_Float16*)take((size_t)NHEADS*LSP*LSP*2);   // per-sample
    _Float16* spvT  = (_Float16*)take((size_t)NS*LSP*CH*2);

    // 1. depthwise conv + BN
    k_dwconv_bn<<<(NS*CH*LSP+255)/256, 256, 0, stream>>>(x, dw_w, bn_g, bn_b, bn_m, bn_v, xd);
    // weight -> f16 copies
    k_f32_to_f16<<<(QKVC*CH+255)/256, 256, 0, stream>>>(w_qkv_ch, wq16c, QKVC*CH);
    k_f32_to_f16<<<(CH*CH+255)/256,   256, 0, stream>>>(w_out_ch, wo16c, CH*CH);
    k_f32_to_f16<<<(QKVC*CH+255)/256, 256, 0, stream>>>(w_qkv_sp, wq16s, QKVC*CH);
    k_f32_to_f16<<<(CH*CH+255)/256,   256, 0, stream>>>(w_out_sp, wo16s, CH*CH);

    // ---- channel attention ----
    k_gn_stats<<<NS, 256, 0, stream>>>(xd, stats);
    k_gn_apply<<<(NS*CH*LSP+255)/256, 256, 0, stream>>>(xd, stats, lnch_g, lnch_b, xnT);
    // 72 M-tiles / 8 waves = 9 groups, 16 N-tiles
    k_gemm_qkv_ch<<<dim3(9*16, NS), 256, 0, stream>>>(wq16c, xnT, qch, kch, vTch);
    // M=384 -> 3 groups, 24 N-tiles
    k_gemm_score<<<dim3(3*24, NS), 256, 0, stream>>>(qch, kch, Sch,
        24, LSP, CH, CH*LSP, CH*LSP, CH*CH);
    k_softmax<<<NS*CH, 128, 0, stream>>>(Sch, attch, CH, 0.0625f);   // 1/sqrt(256)
    k_gemm_av_ch<<<dim3(3*16, NS), 256, 0, stream>>>(attch, vTch, valsT);
    k_gemm_out<<<dim3(3*16, NS), 256, 0, stream>>>(wo16c, valsT, b_out_ch, xd);

    // ---- spatial multi-head attention ----
    k_gn_stats<<<NS, 256, 0, stream>>>(xd, stats);
    k_gn_apply<<<(NS*CH*LSP+255)/256, 256, 0, stream>>>(xd, stats, lnsp_g, lnsp_b, xnT);
    k_gemm_qkv_sp<<<dim3(9*16, NS), 256, 0, stream>>>(wq16s, xnT, qsp, ksp, vsp);
    for (int n = 0; n < NS; ++n) {
        // M=256 -> 2 groups, 16 N-tiles, K=32
        k_gemm_score<<<dim3(2*16, NHEADS), 256, 0, stream>>>(
            qsp + (size_t)n*NHEADS*LSP*HD, ksp + (size_t)n*NHEADS*LSP*HD, Ssp,
            16, HD, LSP, LSP*HD, LSP*HD, LSP*LSP);
        k_softmax<<<NHEADS*LSP, 128, 0, stream>>>(Ssp, attsp, LSP, 0.17677669529663687f);
        // M=256 -> 2 groups, 2 N-tiles, K=256
        k_gemm_av_sp<<<dim3(2*2, NHEADS), 256, 0, stream>>>(
            attsp, vsp + (size_t)n*NHEADS*HD*LSP, spvT + (size_t)n*LSP*CH);
    }
    k_gemm_out<<<dim3(3*16, NS), 256, 0, stream>>>(wo16s, spvT, b_out_sp, xd);

    // ---- upsample + residual ----
    k_upsample<<<(NS*CH*64*64+255)/256, 256, 0, stream>>>(x, xd, up_w, up_b, out);
}
